// Encoder_76295799046485
// MI455X (gfx1250) — matrix operand order repaired and checked by reading, not compiled, not run on hardware
//
#include <hip/hip_runtime.h>

typedef __attribute__((ext_vector_type(16))) _Float16 v16h;
typedef __attribute__((ext_vector_type(8)))  float    v8f;
typedef __attribute__((ext_vector_type(4)))  float    v4f;

#define TT 1000
#define BB 2048
#define II 13
#define HH 13
#define GG 52   /* 4*H */

// ---- fast transcendentals on the native TRANS pipe ----
__device__ __forceinline__ float fast_exp(float x) {
  return __builtin_amdgcn_exp2f(x * 1.44269504088896340736f);   // v_exp_f32
}
__device__ __forceinline__ float fast_sigmoid(float x) {
  return __builtin_amdgcn_rcpf(1.0f + fast_exp(-x));            // v_rcp_f32
}
__device__ __forceinline__ float fast_tanh(float x) {
  // tanh(x) = 1 - 2/(exp(2x)+1); exact saturation at +-inf via rcp(inf)=0
  float e = fast_exp(2.0f * x);
  return 1.0f - 2.0f * __builtin_amdgcn_rcpf(e + 1.0f);
}

__global__ __launch_bounds__(32) void bilstm_wmma_kernel(
    const float* __restrict__ x,
    const float* __restrict__ ln_w, const float* __restrict__ ln_b,
    const float* __restrict__ wih_f, const float* __restrict__ whh_f,
    const float* __restrict__ bih_f, const float* __restrict__ bhh_f,
    const float* __restrict__ wih_b, const float* __restrict__ whh_b,
    const float* __restrict__ bih_b, const float* __restrict__ bhh_b,
    float* __restrict__ out)
{
  const int lane   = threadIdx.x;        // 0..31, wave32
  const int row    = lane & 15;          // batch row within this wave's tile
  const bool hiH   = lane >= 16;         // upper half-wave
  const int d      = blockIdx.y;         // 0 = forward, 1 = backward
  const int b      = blockIdx.x * 16 + row;

  const float* wih = d ? wih_b : wih_f;
  const float* whh = d ? whh_b : whh_f;
  const float* bih = d ? bih_b : bih_f;
  const float* bhh = d ? bhh_b : bhh_f;

  // h staged as f16, row-padded to 16 halfs (32B) so the next-step A-matrix
  // fragment is a single ds_load_b128 per lane. Slots j>=13 stay zero forever.
  __shared__ __align__(16) _Float16 h_lds[16 * 16];
  // gate pre-activations, [gate][row] so accumulator VGPRs store contiguously
  __shared__ __align__(16) float    g_lds[64 * 16];

  {
    unsigned int* p = (unsigned int*)h_lds;           // zero 512B (h0 = 0)
    for (int i = lane; i < 128; i += 32) p[i] = 0u;
  }

  // LayerNorm params: wave-uniform loads (scalarized by the compiler)
  float lw[II], lb[II];
#pragma unroll
  for (int i = 0; i < II; i++) { lw[i] = ln_w[i]; lb[i] = ln_b[i]; }

  // ---- pre-pack W_ih / W_hh into 16-bit WMMA B layout (32x16, K x N) ----
  // lanes 0-15 hold K=0..7 in slots 0..7, lanes 16-31 hold K=8..15 in slots
  // 0..7 (K>=13 zero); slots 8..15 (K=16..31) are zero padding.
  v16h wihB[4], whhB[4];
  float bias4[4];
#pragma unroll
  for (int tn = 0; tn < 4; tn++) {
    v16h wa, wb;
#pragma unroll
    for (int k = 0; k < 16; k++) { wa[k] = (_Float16)0.0f; wb[k] = (_Float16)0.0f; }
    const int gate = tn * 16 + row;
    const bool gv  = (gate < GG);
    if (gv) {
#pragma unroll
      for (int k = 0; k < 8; k++) {
        const int kk = (hiH ? 8 : 0) + k;
        if (kk < II) {
          wa[k] = (_Float16)wih[gate * II + kk];
          wb[k] = (_Float16)whh[gate * II + kk];
        }
      }
    }
    wihB[tn] = wa;
    whhB[tn] = wb;
    bias4[tn] = gv ? (bih[gate] + bhh[gate]) : 0.0f;   // bias depends on N only
  }

  float cst[7], hst[7];
#pragma unroll
  for (int q = 0; q < 7; q++) { cst[q] = 0.0f; hst[q] = 0.0f; }
  const int j0 = hiH ? 7 : 0;                // elementwise j-range split

  const size_t xrow  = (size_t)b * (TT * II);
  const size_t orow  = (size_t)b * (TT * 26) + (size_t)d * 13;

  __syncthreads();                            // h_lds zero visible (S_NOP-cheap)

  for (int s = 0; s < TT; s++) {
    const int t = d ? (TT - 1 - s) : s;
    const float* xp = x + xrow + (size_t)t * II;

    // prefetch next timestep's x row (hide L2 latency in the serial chain)
    if (s + 1 < TT) {
      const float* xnext = x + xrow + (size_t)(d ? t - 1 : t + 1) * II;
      __builtin_prefetch(xnext, 0, 0);        // global_prefetch_b8
    }

    // ---- load x row + LayerNorm (redundant in lane m and lane 16+m) ----
    float xv[II];
#pragma unroll
    for (int i = 0; i < II; i++) xv[i] = xp[i];
    float mu = 0.0f;
#pragma unroll
    for (int i = 0; i < II; i++) mu += xv[i];
    mu *= (1.0f / II);
    float var = 0.0f;
#pragma unroll
    for (int i = 0; i < II; i++) { float dd = xv[i] - mu; var += dd * dd; }
    var *= (1.0f / II);
    const float rs = __builtin_amdgcn_rsqf(var + 1e-5f);
    float xn[II];
#pragma unroll
    for (int i = 0; i < II; i++) xn[i] = (xv[i] - mu) * rs * lw[i] + lb[i];

    // ---- build A fragments (16x32 f16 layout, K=13 zero-padded) ----
    v16h xa;
#pragma unroll
    for (int k = 0; k < 16; k++) xa[k] = (_Float16)0.0f;
    if (!hiH) {
#pragma unroll
      for (int k = 0; k < 8; k++) xa[k] = (_Float16)xn[k];      // K=0..7
    } else {
#pragma unroll
      for (int k = 0; k < 5; k++) xa[k] = (_Float16)xn[8 + k];  // K=8..12
    }

    v16h ha;
#pragma unroll
    for (int k = 0; k < 16; k++) ha[k] = (_Float16)0.0f;
    {
      union { uint4 u; _Float16 h[8]; } cv;
      cv.u = *(const uint4*)((const char*)h_lds + row * 32 + (hiH ? 16 : 0));
#pragma unroll
      for (int k = 0; k < 8; k++) ha[k] = cv.h[k];
    }

    // ---- g = bias + xn*Wih^T + h*Whh^T : 2 chained WMMAs x 4 gate tiles ----
#pragma unroll
    for (int tn = 0; tn < 4; tn++) {
      v8f acc;
#pragma unroll
      for (int r = 0; r < 8; r++) acc[r] = bias4[tn];
      acc = __builtin_amdgcn_wmma_f32_16x16x32_f16(
                false, xa, false, wihB[tn], (short)0, acc, false, false);
      acc = __builtin_amdgcn_wmma_f32_16x16x32_f16(
                false, ha, false, whhB[tn], (short)0, acc, false, false);
      // D layout: lane = N(gate within tile), VGPR r = M(row) (+8 for hi half)
      float* gp = &g_lds[(tn * 16 + row) * 16 + (hiH ? 8 : 0)];
      v4f lo = { acc[0], acc[1], acc[2], acc[3] };
      v4f hi = { acc[4], acc[5], acc[6], acc[7] };
      *(v4f*)gp       = lo;                   // ds_store_b128
      *(v4f*)(gp + 4) = hi;                   // ds_store_b128
    }
    __syncthreads();

    // ---- gates / state update: lanes 0-15 -> j=0..6, lanes 16-31 -> j=7..12
    const size_t ob = orow + (size_t)t * 26;
#pragma unroll
    for (int q = 0; q < 7; q++) {
      const int j = j0 + q;
      if (j < HH) {
        const float iv = fast_sigmoid(g_lds[(     j) * 16 + row]);
        const float fv = fast_sigmoid(g_lds[(13 + j) * 16 + row]);
        const float gv = fast_tanh   (g_lds[(26 + j) * 16 + row]);
        const float ov = fast_sigmoid(g_lds[(39 + j) * 16 + row]);
        const float c2 = fv * cst[q] + iv * gv;
        const float hv = ov * fast_tanh(c2);
        cst[q] = c2;
        hst[q] = hv;
        h_lds[row * 16 + j] = (_Float16)hv;   // f16 stage for next A fragment
        out[ob + j] = hv;                     // hs output
      }
    }
    __syncthreads();
  }

  // ---- final h_n / c_n ----
  const size_t hbase = (size_t)BB * TT * 26;
  const size_t cbase = hbase + (size_t)2 * BB * HH;
#pragma unroll
  for (int q = 0; q < 7; q++) {
    const int j = j0 + q;
    if (j < HH) {
      const size_t off = (size_t)d * BB * HH + (size_t)b * HH + j;
      out[hbase + off] = hst[q];
      out[cbase + off] = cst[q];
    }
  }
}

extern "C" void kernel_launch(void* const* d_in, const int* in_sizes, int n_in,
                              void* d_out, int out_size, void* d_ws, size_t ws_size,
                              hipStream_t stream) {
  (void)in_sizes; (void)n_in; (void)d_ws; (void)ws_size; (void)out_size;
  const float* x     = (const float*)d_in[0];
  const float* ln_w  = (const float*)d_in[1];
  const float* ln_b  = (const float*)d_in[2];
  const float* wih_f = (const float*)d_in[3];
  const float* whh_f = (const float*)d_in[4];
  const float* bih_f = (const float*)d_in[5];
  const float* bhh_f = (const float*)d_in[6];
  const float* wih_b = (const float*)d_in[7];
  const float* whh_b = (const float*)d_in[8];
  const float* bih_b = (const float*)d_in[9];
  const float* bhh_b = (const float*)d_in[10];
  float* out = (float*)d_out;

  dim3 grid(BB / 16, 2);   // 128 batch tiles x 2 directions = 256 waves
  dim3 block(32);          // one wave32 per workgroup; barriers degrade to S_NOP
  bilstm_wmma_kernel<<<grid, block, 0, stream>>>(
      x, ln_w, ln_b, wih_f, whh_f, bih_f, bhh_f,
      wih_b, whh_b, bih_b, bhh_b, out);
}
